// GNCA_56968446214209
// MI455X (gfx1250) — matrix operand order repaired and checked by reading, not compiled, not run on hardware
//
#include <hip/hip_runtime.h>
#include <hip/hip_bf16.h>
#include <stdint.h>

// ---- CDNA5 (gfx1250) wave32 WMMA GNN layer ----
typedef __attribute__((ext_vector_type(16))) __bf16        v16bf;
typedef __attribute__((ext_vector_type(8)))  float         v8f;
typedef __attribute__((ext_vector_type(4)))  unsigned int  v4u;
typedef __attribute__((ext_vector_type(4)))  float         v4f;

#define TM   32      // node rows per block
#define SWP  136     // LDS weight tile row stride (128 + 8 pad)
#define SHP  264     // LDS hidden row stride (256 + 8 pad)

// 1 = use gfx1250 GLOBAL_LOAD_ASYNC_TO_LDS_B128 for weight tiles (ASYNCcnt path)
#define USE_ASYNC_LDS 1

union FragU { v16bf v; v4u q[2]; };

// A-matrix fragment (16x32 bf16), ISA 7.12.2 layout:
// lanes 0-15: M=lane, K = {0..7, 16..23}; lanes 16-31: M=lane-16, K = {8..15, 24..31}
__device__ inline v16bf load_a(const __bf16* S, int stride, int row0, int k0, int lo, int hi) {
  FragU f;
  const __bf16* p = S + (size_t)(row0 + lo) * stride + k0 + hi * 8;
  f.q[0] = *(const v4u*)(p);        // K offsets 0..7
  f.q[1] = *(const v4u*)(p + 16);   // K offsets 16..23
  return f.v;
}
// B-matrix fragment (32x16 bf16): lanes 0-15 hold K = {0..7, 16..23} for N=lane, lanes 16-31 hold K = {8..15, 24..31}
__device__ inline v16bf load_b(const __bf16* S, int stride, int n0, int k0, int lo, int hi) {
  FragU f;
  const __bf16* p = S + (size_t)(n0 + lo) * stride + k0 + hi * 8;
  f.q[0] = *(const v4u*)(p);        // K offsets 0..7
  f.q[1] = *(const v4u*)(p + 16);   // K offsets 16..23
  return f.v;
}

// Fill a 64x128 bf16 weight tile in LDS from pre-converted bf16 weights.
// Each of 256 threads moves 4 x 16B chunks. Async path: global->LDS DMA,
// per-wave wait on ASYNCcnt before the workgroup barrier at the call site.
__device__ inline void load_w_tile(__bf16* sW, const __bf16* __restrict__ Wb,
                                   int nbase, int kbase, int rowlen, int tid) {
#pragma unroll
  for (int i = 0; i < 4; ++i) {
    int chunk = tid + i * 256;          // 0..1023
    int r = chunk >> 4;                 // 0..63
    int c = (chunk & 15) * 8;           // 0..120 (bf16 elems)
#if USE_ASYNC_LDS
    unsigned goff = (unsigned)(((unsigned)(nbase + r) * (unsigned)rowlen + (unsigned)(kbase + c)) * 2u);
    unsigned lds  = (unsigned)(size_t)(&sW[r * SWP + c]);   // low 32 bits = LDS byte address
    asm volatile("global_load_async_to_lds_b128 %0, %1, %2"
                 :: "v"(lds), "v"(goff), "s"((unsigned long long)(size_t)Wb)
                 : "memory");
#else
    *(v4u*)&sW[r * SWP + c] = *(const v4u*)&Wb[(size_t)(nbase + r) * rowlen + kbase + c];
#endif
  }
#if USE_ASYNC_LDS
  asm volatile("s_wait_asynccnt 0x0" ::: "memory");
#endif
}

// Generic 2-layer MLP over 32-row node tiles:
//   H = relu(X @ W1^T + b1)   (X: [TM x K1], W1: [256 x K1], bf16)
//   O = H @ W2^T + b2 (+resid)  (W2: [128 x 256], bf16)
template<int K1>
__global__ __launch_bounds__(256) void mlp_kernel(
    const float* __restrict__ Xa, const float* __restrict__ Xb,
    const __bf16* __restrict__ W1, const float* __restrict__ b1,
    const __bf16* __restrict__ W2, const float* __restrict__ b2,
    const float* __restrict__ resid, float* __restrict__ Out, int Nn)
{
  constexpr int SXP = K1 + 8;
  __shared__ __align__(16) __bf16 sX[TM * SXP];
  __shared__ __align__(16) __bf16 sH[TM * SHP];
  __shared__ __align__(16) __bf16 sW[64 * SWP];

  const int tid   = threadIdx.x;
  const int w     = tid >> 5;
  const int lane  = tid & 31;
  const int lo    = lane & 15;
  const int hi    = lane >> 4;
  const int mtile = (w >> 2) * 16;   // 0 or 16
  const int ntile = (w & 3) * 16;    // 0..48 inside a 64-wide N chunk
  const int m0    = blockIdx.x * TM;

  // ---- stage 0: load X tile (optionally concat [Xa||Xb]) as bf16 ----
  for (int idx = tid; idx < TM * K1; idx += 256) {
    int r = idx / K1, c = idx % K1;
    int g = m0 + r; if (g >= Nn) g = Nn - 1;      // clamp (stores are guarded)
    float v;
    if (K1 == 128) v = Xa[(size_t)g * 128 + c];
    else           v = (c < 128) ? Xa[(size_t)g * 128 + c]
                                 : Xb[(size_t)g * 128 + (c - 128)];
    sX[r * SXP + c] = (__bf16)v;
  }

  // ---- stage 1: sH = relu(sX @ W1^T + b1), H = 256 in four 64-wide chunks ----
  for (int nc = 0; nc < 4; ++nc) {
    v8f acc = {0.f, 0.f, 0.f, 0.f, 0.f, 0.f, 0.f, 0.f};
    for (int kc = 0; kc < K1 / 128; ++kc) {
      __syncthreads();
      load_w_tile(sW, W1, nc * 64, kc * 128, K1, tid);
      __syncthreads();
#pragma unroll
      for (int ks = 0; ks < 4; ++ks) {
        v16bf a = load_a(sX, SXP, mtile, kc * 128 + ks * 32, lo, hi);
        v16bf b = load_b(sW, SWP, ntile, ks * 32, lo, hi);
        acc = __builtin_amdgcn_wmma_f32_16x16x32_bf16(
                  false, a, false, b, (short)0, acc, false, false);
      }
    }
    const int col  = nc * 64 + ntile + lo;
    const float bs = b1[col];
#pragma unroll
    for (int j = 0; j < 8; ++j) {
      float v = acc[j] + bs;
      v = v > 0.f ? v : 0.f;
      sH[(mtile + hi * 8 + j) * SHP + col] = (__bf16)v;
    }
  }

  // ---- stage 2: O = sH @ W2^T + b2 (+resid), D = 128 in two 64-wide chunks ----
  for (int nc = 0; nc < 2; ++nc) {
    v8f acc = {0.f, 0.f, 0.f, 0.f, 0.f, 0.f, 0.f, 0.f};
    for (int kc = 0; kc < 2; ++kc) {
      __syncthreads();
      load_w_tile(sW, W2, nc * 64, kc * 128, 256, tid);
      __syncthreads();
#pragma unroll
      for (int ks = 0; ks < 4; ++ks) {
        v16bf a = load_a(sH, SHP, mtile, kc * 128 + ks * 32, lo, hi);
        v16bf b = load_b(sW, SWP, ntile, ks * 32, lo, hi);
        acc = __builtin_amdgcn_wmma_f32_16x16x32_bf16(
                  false, a, false, b, (short)0, acc, false, false);
      }
    }
    const int col  = nc * 64 + ntile + lo;
    const float bs = b2[col];
#pragma unroll
    for (int j = 0; j < 8; ++j) {
      int grow = m0 + mtile + hi * 8 + j;
      if (grow < Nn) {
        float v = acc[j] + bs;
        if (resid) v += resid[(size_t)grow * 128 + col];
        Out[(size_t)grow * 128 + col] = v;
      }
    }
  }
}

// one-shot f32 -> bf16 weight conversion (weights stay hot in L2)
__global__ __launch_bounds__(256) void cvt_bf16_kernel(const float* __restrict__ s,
                                                       __bf16* __restrict__ d, int n) {
  int i = blockIdx.x * 256 + threadIdx.x;
  if (i < n) d[i] = (__bf16)s[i];
}

__global__ __launch_bounds__(256) void zero_kernel(v4f* __restrict__ p, long n4) {
  long i = (long)blockIdx.x * blockDim.x + threadIdx.x;
  long st = (long)gridDim.x * blockDim.x;
  v4f z = {0.f, 0.f, 0.f, 0.f};
  for (; i < n4; i += st) p[i] = z;
}

// one wave per edge: lane grabs a float4 of m[sender], 4 HW f32 atomics into agg[receiver]
__global__ __launch_bounds__(256) void scatter_kernel(
    const float* __restrict__ m, const int* __restrict__ snd,
    const int* __restrict__ rcv, float* __restrict__ agg, int E)
{
  int gw = (int)((blockIdx.x * 256u + threadIdx.x) >> 5);
  int lane = threadIdx.x & 31;
  if (gw >= E) return;
  int s = snd[gw], r = rcv[gw];
  v4f v = ((const v4f*)(m + (size_t)s * 128))[lane];
  float* d = agg + (size_t)r * 128 + lane * 4;
  unsafeAtomicAdd(d + 0, v.x);
  unsafeAtomicAdd(d + 1, v.y);
  unsafeAtomicAdd(d + 2, v.z);
  unsafeAtomicAdd(d + 3, v.w);
}

__global__ __launch_bounds__(256) void edge_kernel(const v4f* __restrict__ e,
                                                   v4f* __restrict__ o, long n4) {
  long i = (long)blockIdx.x * blockDim.x + threadIdx.x;
  long st = (long)gridDim.x * blockDim.x;
  for (; i < n4; i += st) { v4f x = e[i]; o[i] = x + x; }
}

extern "C" void kernel_launch(void* const* d_in, const int* in_sizes, int n_in,
                              void* d_out, int out_size, void* d_ws, size_t ws_size,
                              hipStream_t stream) {
  const float* nodes     = (const float*)d_in[0];
  const float* edges     = (const float*)d_in[1];
  const int*   senders   = (const int*)d_in[2];
  const int*   receivers = (const int*)d_in[3];
  const float* Wm1 = (const float*)d_in[4];
  const float* bm1 = (const float*)d_in[5];
  const float* Wm2 = (const float*)d_in[6];
  const float* bm2 = (const float*)d_in[7];
  const float* Wn1 = (const float*)d_in[8];
  const float* bn1 = (const float*)d_in[9];
  const float* Wn2 = (const float*)d_in[10];
  const float* bn2 = (const float*)d_in[11];

  const int N = in_sizes[0] / 128;
  const int E = in_sizes[2];

  float* out_nodes = (float*)d_out;
  float* out_edges = out_nodes + (size_t)N * 128;
  // temp m lives in the edge-output region (E*16 >= N*128 floats);
  // it is fully rewritten each call before use, and the edge kernel runs last.
  float* m   = out_edges;

  // workspace: [ agg: N*128 f32 | bf16 weights: Wm1,Wm2,Wn1,Wn2 ]
  float*  agg  = (float*)d_ws;
  __bf16* Wm1b = (__bf16*)(agg + (size_t)N * 128);
  __bf16* Wm2b = Wm1b + 256 * 128;
  __bf16* Wn1b = Wm2b + 128 * 256;
  __bf16* Wn2b = Wn1b + 256 * 256;

  // 0) one-shot weight conversion to bf16 (stays resident in L2)
  cvt_bf16_kernel<<<128, 256, 0, stream>>>(Wm1, Wm1b, 256 * 128);
  cvt_bf16_kernel<<<128, 256, 0, stream>>>(Wm2, Wm2b, 128 * 256);
  cvt_bf16_kernel<<<256, 256, 0, stream>>>(Wn1, Wn1b, 256 * 256);
  cvt_bf16_kernel<<<128, 256, 0, stream>>>(Wn2, Wn2b, 128 * 256);

  const int blocksM = (N + TM - 1) / TM;

  // 1) m = message_mlp(nodes)
  mlp_kernel<128><<<blocksM, 256, 0, stream>>>(nodes, nullptr, Wm1b, bm1, Wm2b, bm2,
                                               nullptr, m, N);
  // 2) agg = 0 ; agg[receivers] += m[senders]
  long n4 = (long)N * 128 / 4;
  zero_kernel<<<2048, 256, 0, stream>>>((v4f*)agg, n4);
  int blocksS = (int)(((long)E * 32 + 255) / 256);
  scatter_kernel<<<blocksS, 256, 0, stream>>>(m, senders, receivers, agg, E);
  // 3) out_nodes = nodes + node_mlp([nodes || agg])
  mlp_kernel<256><<<blocksM, 256, 0, stream>>>(nodes, agg, Wn1b, bn1, Wn2b, bn2,
                                               nodes, out_nodes, N);
  // 4) out_edges = 2 * edges  (after scatter has consumed m)
  long e4 = (long)E * 16 / 4;
  edge_kernel<<<2048, 256, 0, stream>>>((const v4f*)edges, (v4f*)out_edges, e4);
}
